// MyGNN_44856638439998
// MI455X (gfx1250) — matrix operand order repaired and checked by reading, not compiled, not run on hardware
//
#include <hip/hip_runtime.h>
#include <stdint.h>

// GATConv-style fused pipeline for MI455X (gfx1250, wave32, WMMA).
//   x1 = data @ W.T                         (WMMA f16->f32 GEMM)
//   a_s = x1@att_src ; a_d = x1@att_dst
//   m_j = lrelu(prefixmax(a_s)[j] + a_d[j]) (lrelu monotone => exact)
//   Z_j = sum_{i<=j} exp(lrelu(a_s[i]+a_d[j]) - m_j)
//   out = relu( (1/Z) * sum_{i<=j} exp(...) * x1[i,:] + bias )   (fused WMMA GEMM)
//
// Data-movement decisions (roofline, 23.3 TB/s HBM / 192 MB L2):
//  * Never materialize the 64 MB alpha matrix: weights are regenerated per
//    K-tile from two 16 KB vectors directly in the WMMA A-operand layout.
//  * Kernel 1 stores the f16 copy of x1 TRANSPOSED (x1hT[c][i]) so the WMMA
//    B operand is one aligned 32B load per lane (L2-resident, no LDS tile,
//    no barriers in the dominant K loop).
//  * Each consumer wave covers 32 dst rows x 64 channels: every B fragment
//    feeds two WMMAs, halving L2 B-traffic per FLOP.
//  * a_s is DMA'd once per block into LDS via global_load_async_to_lds_b128
//    (ASYNCcnt) and read with aligned ds_load_b128 in the K loop.

#define NN        4096
#define CIN       512
#define COUTC     256
#define NEG_SLOPE 0.2f

typedef __attribute__((ext_vector_type(16))) _Float16 v16h;
typedef __attribute__((ext_vector_type(8)))  _Float16 v8h;
typedef __attribute__((ext_vector_type(8)))  float    v8f;
typedef __attribute__((ext_vector_type(4)))  float    v4f;

static __device__ __forceinline__ float lrelu(float x) {
    return x > 0.f ? x : NEG_SLOPE * x;
}

// ---------------------------------------------------------------------------
// Kernel 1: x1 = data @ W.T   (M=4096, K=512, N=256), one 16x16 tile per wave.
// A layout (f16 16x32): lane<16 row M=lane holds K {kk..kk+7, kk+16..kk+23};
// lane>=16 row M=lane-16 holds K {kk+8..kk+15, kk+24..kk+31}.
// B layout (f16 32x16): col N=lane%16, same K split per lane half as A.
// Writes x1f row-major (for the logit dot products) and x1hT transposed f16
// (one aligned b128 store per lane).
// ---------------------------------------------------------------------------
__global__ void __launch_bounds__(256)
k_gemm_x1(const float* __restrict__ data, const float* __restrict__ W,
          float* __restrict__ x1f, _Float16* __restrict__ x1hT)
{
    const int lane = threadIdx.x & 31;
    const int wave = threadIdx.x >> 5;
    const int tile = blockIdx.x * 8 + wave;   // 4096 tiles total
    const int mT = tile >> 4;                 // 256 row tiles
    const int nT = tile & 15;                 // 16 col tiles
    const int r0 = mT * 16, c0 = nT * 16;
    const int lh = lane & 15, hs = lane >> 4;

    const float* arow = data + (size_t)(r0 + lh) * CIN;  // A row for this lane
    const float* brow = W    + (size_t)(c0 + lh) * CIN;  // W row = B column

    v8f acc = {};
    for (int kk = 0; kk < CIN; kk += 32) {
        const int ka = kk + hs * 8;
        v4f a0 = *(const v4f*)(arow + ka);
        v4f a1 = *(const v4f*)(arow + ka + 4);
        v4f a2 = *(const v4f*)(arow + ka + 16);
        v4f a3 = *(const v4f*)(arow + ka + 20);
        const int kb = kk + hs * 8;
        v4f b0 = *(const v4f*)(brow + kb);
        v4f b1 = *(const v4f*)(brow + kb + 4);
        v4f b2 = *(const v4f*)(brow + kb + 16);
        v4f b3 = *(const v4f*)(brow + kb + 20);
        v16h a, b;
        #pragma unroll
        for (int e = 0; e < 4; ++e) {
            a[e]      = (_Float16)a0[e];
            a[4 + e]  = (_Float16)a1[e];
            a[8 + e]  = (_Float16)a2[e];
            a[12 + e] = (_Float16)a3[e];
            b[e]      = (_Float16)b0[e];
            b[4 + e]  = (_Float16)b1[e];
            b[8 + e]  = (_Float16)b2[e];
            b[12 + e] = (_Float16)b3[e];
        }
        acc = __builtin_amdgcn_wmma_f32_16x16x32_f16(false, a, false, b,
                                                     (short)0, acc, false, false);
    }
    // C/D layout: VGPR v, lane<16 -> (row v, col lane); lane>=16 -> (row v+8, col lane-16)
    const int c = c0 + lh;
    v8h hv;
    #pragma unroll
    for (int v = 0; v < 8; ++v) {
        const int r = r0 + v + hs * 8;
        x1f[(size_t)r * COUTC + c] = acc[v];
        hv[v] = (_Float16)acc[v];
    }
    // transposed f16 copy: 8 consecutive rows for fixed column -> one b128 store
    *(v8h*)(x1hT + (size_t)c * NN + r0 + hs * 8) = hv;
}

// ---------------------------------------------------------------------------
// Kernel 2: attention logits a_s[n] = x1[n,:]@att_src, a_d likewise.
// One wave per row, lane covers 8 channels, shuffle reduce (wave32).
// ---------------------------------------------------------------------------
__global__ void __launch_bounds__(256)
k_logits(const float* __restrict__ x1f, const float* __restrict__ att_src,
         const float* __restrict__ att_dst,
         float* __restrict__ a_s, float* __restrict__ a_d)
{
    const int lane = threadIdx.x & 31;
    const int wave = threadIdx.x >> 5;
    const int row  = blockIdx.x * 8 + wave;
    const float* xr = x1f + (size_t)row * COUTC + lane * 8;
    v4f x0  = *(const v4f*)xr;
    v4f x1v = *(const v4f*)(xr + 4);
    v4f as0 = *(const v4f*)(att_src + lane * 8);
    v4f as1 = *(const v4f*)(att_src + lane * 8 + 4);
    v4f ad0 = *(const v4f*)(att_dst + lane * 8);
    v4f ad1 = *(const v4f*)(att_dst + lane * 8 + 4);
    float ss = 0.f, sd = 0.f;
    #pragma unroll
    for (int e = 0; e < 4; ++e) {
        ss += x0[e] * as0[e] + x1v[e] * as1[e];
        sd += x0[e] * ad0[e] + x1v[e] * ad1[e];
    }
    #pragma unroll
    for (int off = 16; off > 0; off >>= 1) {
        ss += __shfl_down(ss, off);
        sd += __shfl_down(sd, off);
    }
    if (lane == 0) { a_s[row] = ss; a_d[row] = sd; }
}

// ---------------------------------------------------------------------------
// Kernel 3: inclusive prefix max of a_s (single wave, 128 elems per lane).
// ---------------------------------------------------------------------------
__global__ void __launch_bounds__(32)
k_scan(const float* __restrict__ a_s, float* __restrict__ pmax)
{
    const int lane = threadIdx.x;
    const int base = lane * 128;
    float m = -3.4e38f;
    for (int t = 0; t < 128; ++t) m = fmaxf(m, a_s[base + t]);
    float sc = m;
    #pragma unroll
    for (int off = 1; off < 32; off <<= 1) {
        float v = __shfl_up(sc, off);
        if (lane >= off) sc = fmaxf(sc, v);
    }
    float excl = __shfl_up(sc, 1);
    float run = (lane == 0) ? -3.4e38f : excl;
    for (int t = 0; t < 128; ++t) {
        run = fmaxf(run, a_s[base + t]);
        pmax[base + t] = run;
    }
}

// ---------------------------------------------------------------------------
// Kernel 4: per-destination softmax stats. One wave per column j.
// m_j = lrelu(pmax[j] + a_d[j]); Z_j = sum_{i<=j} exp(lrelu(a_s[i]+a_d[j]) - m_j)
// ---------------------------------------------------------------------------
__global__ void __launch_bounds__(256)
k_stats(const float* __restrict__ a_s, const float* __restrict__ a_d,
        const float* __restrict__ pmax,
        float* __restrict__ mj, float* __restrict__ zinv)
{
    const int lane = threadIdx.x & 31;
    const int wave = threadIdx.x >> 5;
    const int j = blockIdx.x * 8 + wave;
    const float adj = a_d[j];
    const float m  = lrelu(pmax[j] + adj);
    float z = 0.f;
    for (int i = lane; i <= j; i += 32) {
        z += __expf(lrelu(a_s[i] + adj) - m);
    }
    #pragma unroll
    for (int off = 16; off > 0; off >>= 1) z += __shfl_down(z, off);
    if (lane == 0) { mj[j] = m; zinv[j] = 1.f / z; }
}

// ---------------------------------------------------------------------------
// Kernel 5: fused masked-softmax aggregation  out = relu(alpha.T @ x1 + bias).
// One wave per (32 dst rows x 64 channels) work unit: two A fragments
// (lower/upper 16-row j-tiles, regenerated softmax weights) x four B
// sub-tiles -> 8 WMMAs per K step, each B fragment reused twice. B fragments
// are single aligned v16h loads from the transposed f16 x1 (L2 resident).
// a_s lives in LDS (async DMA once per block). No barriers in the K loop.
// ---------------------------------------------------------------------------
__global__ void __launch_bounds__(256)
k_att_gemm(const _Float16* __restrict__ x1hT,
           const float* __restrict__ a_s, const float* __restrict__ a_d,
           const float* __restrict__ mj, const float* __restrict__ zinv,
           const float* __restrict__ bias, float* __restrict__ out)
{
    __shared__ float lds_as[NN];          // 16 KB: full source-logit vector

    const int tid  = threadIdx.x;
    const int lane = tid & 31;
    const int wave = tid >> 5;

    // ---- async DMA of a_s into LDS (4 x b128 per thread, ASYNCcnt) ----
    {
        const float* gsrc = a_s + tid * 16;
        // low 32 bits of a flat shared-aperture address = LDS byte offset
        uint32_t loff = (uint32_t)(uintptr_t)(lds_as + tid * 16);
        asm volatile("global_load_async_to_lds_b128 %0, %1, off"
                     :: "v"(loff), "v"(gsrc) : "memory");
        asm volatile("global_load_async_to_lds_b128 %0, %1, off offset:16"
                     :: "v"(loff), "v"(gsrc) : "memory");
        asm volatile("global_load_async_to_lds_b128 %0, %1, off offset:32"
                     :: "v"(loff), "v"(gsrc) : "memory");
        asm volatile("global_load_async_to_lds_b128 %0, %1, off offset:48"
                     :: "v"(loff), "v"(gsrc) : "memory");
        asm volatile("s_wait_asynccnt 0x0" ::: "memory");
    }
    __syncthreads();

    const int wu = blockIdx.x * 8 + wave;   // 512 work units
    const int jP = wu >> 2;                 // 128 dst row-pairs (32 rows each)
    const int cg = wu & 3;                  // 4 channel groups of 64
    const int j0 = jP * 32;
    const int nBase = cg * 64;
    const int lh = lane & 15, hs = lane >> 4;

    const int   jrowA = j0 + lh;            // lower 16-row tile (A-operand M)
    const int   jrowB = j0 + 16 + lh;       // upper 16-row tile
    const float adjA  = a_d[jrowA], mjlA = mj[jrowA];
    const float adjB  = a_d[jrowB], mjlB = mj[jrowB];

    // per-lane base pointers for the 4 B-operand column sub-tiles
    const _Float16* bcol0 = x1hT + (size_t)(nBase + lh)      * NN + hs * 16;
    const _Float16* bcol1 = x1hT + (size_t)(nBase + 16 + lh) * NN + hs * 16;
    const _Float16* bcol2 = x1hT + (size_t)(nBase + 32 + lh) * NN + hs * 16;
    const _Float16* bcol3 = x1hT + (size_t)(nBase + 48 + lh) * NN + hs * 16;

    v8f accA0 = {}, accA1 = {}, accA2 = {}, accA3 = {};
    v8f accB0 = {}, accB1 = {}, accB2 = {}, accB3 = {};
    const int nChunks = (j0 + 63) >> 5;     // cover i in [0, j0+31]

    for (int ch = 0; ch < nChunks; ++ch) {
        const int iBase = ch * 32;

        // ---- shared source logits for this K tile (aligned LDS reads) ----
        const int kb8 = hs * 16;
        v4f s0 = *(const v4f*)(lds_as + iBase + kb8);
        v4f s1 = *(const v4f*)(lds_as + iBase + kb8 + 4);
        v4f s2 = *(const v4f*)(lds_as + iBase + kb8 + 8);
        v4f s3 = *(const v4f*)(lds_as + iBase + kb8 + 12);
        float sv[16];
        #pragma unroll
        for (int e = 0; e < 4; ++e) {
            sv[e] = s0[e]; sv[4 + e] = s1[e]; sv[8 + e] = s2[e]; sv[12 + e] = s3[e];
        }

        // ---- two A fragments: softmax weights generated on the fly ----
        v16h aA, aB;
        #pragma unroll
        for (int t = 0; t < 16; ++t) {
            const int k = kb8 + t;
            const int i = iBase + k;
            const float scA = lrelu(sv[t] + adjA);
            const float scB = lrelu(sv[t] + adjB);
            const float wA  = (i <= jrowA) ? __expf(scA - mjlA) : 0.f;
            const float wB  = (i <= jrowB) ? __expf(scB - mjlB) : 0.f;
            aA[t] = (_Float16)wA;
            aB[t] = (_Float16)wB;
        }

        // ---- B fragments: one aligned 32B load per lane per sub-tile ----
        v16h b0 = *(const v16h*)(bcol0 + iBase);
        v16h b1 = *(const v16h*)(bcol1 + iBase);
        v16h b2 = *(const v16h*)(bcol2 + iBase);
        v16h b3 = *(const v16h*)(bcol3 + iBase);

        if (iBase + 32 < NN)  // speculative prefetch of next K tile
            __builtin_prefetch(bcol0 + iBase + 32, 0, 0);

        accA0 = __builtin_amdgcn_wmma_f32_16x16x32_f16(false, aA, false, b0,
                                                       (short)0, accA0, false, false);
        accB0 = __builtin_amdgcn_wmma_f32_16x16x32_f16(false, aB, false, b0,
                                                       (short)0, accB0, false, false);
        accA1 = __builtin_amdgcn_wmma_f32_16x16x32_f16(false, aA, false, b1,
                                                       (short)0, accA1, false, false);
        accB1 = __builtin_amdgcn_wmma_f32_16x16x32_f16(false, aB, false, b1,
                                                       (short)0, accB1, false, false);
        accA2 = __builtin_amdgcn_wmma_f32_16x16x32_f16(false, aA, false, b2,
                                                       (short)0, accA2, false, false);
        accB2 = __builtin_amdgcn_wmma_f32_16x16x32_f16(false, aB, false, b2,
                                                       (short)0, accB2, false, false);
        accA3 = __builtin_amdgcn_wmma_f32_16x16x32_f16(false, aA, false, b3,
                                                       (short)0, accA3, false, false);
        accB3 = __builtin_amdgcn_wmma_f32_16x16x32_f16(false, aB, false, b3,
                                                       (short)0, accB3, false, false);
    }

    // ---- epilogue: 1/Z scaling, bias, relu ----
    const float bias0 = bias[nBase + lh];
    const float bias1 = bias[nBase + 16 + lh];
    const float bias2 = bias[nBase + 32 + lh];
    const float bias3 = bias[nBase + 48 + lh];
    #pragma unroll
    for (int v = 0; v < 8; ++v) {
        const int jA = j0 + v + hs * 8;
        const int jB = jA + 16;
        const float ziA = zinv[jA];
        const float ziB = zinv[jB];
        float* orowA = out + (size_t)jA * COUTC + nBase + lh;
        float* orowB = out + (size_t)jB * COUTC + nBase + lh;
        orowA[0]  = fmaxf(accA0[v] * ziA + bias0, 0.f);
        orowA[16] = fmaxf(accA1[v] * ziA + bias1, 0.f);
        orowA[32] = fmaxf(accA2[v] * ziA + bias2, 0.f);
        orowA[48] = fmaxf(accA3[v] * ziA + bias3, 0.f);
        orowB[0]  = fmaxf(accB0[v] * ziB + bias0, 0.f);
        orowB[16] = fmaxf(accB1[v] * ziB + bias1, 0.f);
        orowB[32] = fmaxf(accB2[v] * ziB + bias2, 0.f);
        orowB[48] = fmaxf(accB3[v] * ziB + bias3, 0.f);
    }
}

// ---------------------------------------------------------------------------
extern "C" void kernel_launch(void* const* d_in, const int* in_sizes, int n_in,
                              void* d_out, int out_size, void* d_ws, size_t ws_size,
                              hipStream_t stream)
{
    const float* data    = (const float*)d_in[0];
    const float* W       = (const float*)d_in[1];
    const float* att_src = (const float*)d_in[2];
    const float* att_dst = (const float*)d_in[3];
    const float* bias    = (const float*)d_in[4];
    float* out = (float*)d_out;

    char* ws = (char*)d_ws;
    float*    x1f  = (float*)ws;    ws += (size_t)NN * COUTC * sizeof(float);
    _Float16* x1hT = (_Float16*)ws; ws += (size_t)NN * COUTC * sizeof(_Float16);
    float*    a_s  = (float*)ws;    ws += NN * sizeof(float);
    float*    a_d  = (float*)ws;    ws += NN * sizeof(float);
    float*    pmax = (float*)ws;    ws += NN * sizeof(float);
    float*    mj   = (float*)ws;    ws += NN * sizeof(float);
    float*    zinv = (float*)ws;    ws += NN * sizeof(float);

    k_gemm_x1<<<512, 256, 0, stream>>>(data, W, x1f, x1hT);
    k_logits<<<512, 256, 0, stream>>>(x1f, att_src, att_dst, a_s, a_d);
    k_scan<<<1, 32, 0, stream>>>(a_s, pmax);
    k_stats<<<512, 256, 0, stream>>>(a_s, a_d, pmax, mj, zinv);
    k_att_gemm<<<64, 256, 0, stream>>>(x1hT, a_s, a_d, mj, zinv, bias, out);
}
